// MoELayer_34445637714412
// MI455X (gfx1250) — matrix operand order repaired and checked by reading, not compiled, not run on hardware
//
#include <hip/hip_runtime.h>
#include <hip/hip_bf16.h>

// Problem constants (match reference)
#define ND 1024        // model dim D
#define NH 2048        // hidden dim H
#define NE 8           // experts
#define NTOK 8192      // B*S tokens
#define CAP  NTOK      // worst-case tokens per expert list
#define NENT (2*NTOK)  // total (token, k) entries, entry id = 2*t + k

#define TPB    256     // 8 wave32s
#define NWAVES 8

typedef __attribute__((ext_vector_type(16))) __bf16 v16bf;
typedef __attribute__((ext_vector_type(8)))  float  v8f;

__device__ __forceinline__ v8f wmma_bf16(v16bf a, v16bf b, v8f c) {
    return __builtin_amdgcn_wmma_f32_16x16x32_bf16(
        false, a, false, b, (short)0, c, false, false);
}
__device__ __forceinline__ float silu_mul(float u, float v) {
    return u * __builtin_amdgcn_rcpf(1.f + __expf(-u)) * v;
}

// ---- packed expert kernel tiling: 64 tokens, H processed in two halves ----
#define MT     64
#define HHALF  (NH/2)                       // 1024
#define LDA2   (ND + 8)                     // bf16 elems; row = 2064B (16B aligned)
#define LDG2   (HHALF + 8)
#define SA2_BYTES ((size_t)MT * LDA2 * 2)   // 132096
#define SG2_BYTES ((size_t)MT * LDG2 * 2)   // 132096
#define SMEM2 (SA2_BYTES + SG2_BYTES + (size_t)MT*4)          // ~264.5 KB

// ---- fallback kernel tiling (round-1 style) ----
#define MTF    32
#define LDAF   (ND + 8)
#define LDGF   (NH + 8)
#define SAF_BYTES ((size_t)MTF * LDAF * 2)
#define SGF_BYTES ((size_t)MTF * LDGF * 2)
#define SMEMF (SAF_BYTES + SGF_BYTES + (size_t)MTF*4 + (size_t)MTF*4)

// ---- workspace layout (all offsets multiples of 256) ----
#define OFF_COUNTS 0
#define OFF_ENT    256                                    // NE*CAP*4      = 256 KB
#define OFF_WTE    (256 + (size_t)NE*CAP*4)               // NENT*4        = 64 KB
#define OFF_XB     (OFF_WTE + (size_t)NENT*4)             // NTOK*ND*2     = 16.8 MB
#define OFF_W1P    (OFF_XB  + (size_t)NTOK*ND*2)          // NE*ND*NH*2    = 33.5 MB
#define OFF_W2P    (OFF_W1P + (size_t)NE*ND*NH*2)
#define OFF_W3P    (OFF_W2P + (size_t)NE*ND*NH*2)
#define OFF_CONTRIB (OFF_W3P + (size_t)NE*ND*NH*2)        // NENT*ND*4     = 67 MB
#define WS_NEEDED  (OFF_CONTRIB + (size_t)NENT*ND*4)      // ~176 MB

// ---------------------------------------------------------------------------
// zero output (fallback needs it) + routing counters
// ---------------------------------------------------------------------------
__global__ void moe_init_kernel(float4* __restrict__ out4, int* __restrict__ counts, int n4) {
    int i = blockIdx.x * blockDim.x + threadIdx.x;
    if (i < n4) out4[i] = float4{0.f, 0.f, 0.f, 0.f};
    if (i < NE) counts[i] = 0;
}

// ---------------------------------------------------------------------------
// x fp32 -> bf16
// ---------------------------------------------------------------------------
__global__ void pack_x_kernel(const float4* __restrict__ x4, __bf16* __restrict__ xb) {
    int i = blockIdx.x * blockDim.x + threadIdx.x;
    if (i >= NTOK * ND / 4) return;
    float4 v = x4[i];
    __bf16* o = xb + (size_t)i * 4;
    o[0] = (__bf16)v.x; o[1] = (__bf16)v.y; o[2] = (__bf16)v.z; o[3] = (__bf16)v.w;
}

// ---------------------------------------------------------------------------
// pack weight [E][K][N] fp32 -> fragment-major bf16 tiles.
// Tile = 16-wide n-strip x 32-deep k-chunk; lane l (n=l&15, hi=l>>4) owns
// 16 contiguous bf16 with K = kc*32 + hi*16 + i.  dst index == tid*16 + i.
// ---------------------------------------------------------------------------
__global__ void pack_b_kernel(const float* __restrict__ src, __bf16* __restrict__ dst,
                              int K, int N) {
    const int tid = blockIdx.x * blockDim.x + threadIdx.x;
    const int kchunks = K >> 5;
    const int total = NE * (N >> 4) * kchunks * 32;
    if (tid >= total) return;
    const int lane = tid & 31;
    int rest = tid >> 5;
    const int kc = rest % kchunks; rest /= kchunks;
    const int s  = rest % (N >> 4);
    const int e  = rest / (N >> 4);
    const float* se = src + (size_t)e * K * N;
    const int n     = s * 16 + (lane & 15);
    const int kbase = kc * 32 + (lane >> 4) * 16;
    v16bf v;
#pragma unroll
    for (int i = 0; i < 16; ++i) v[i] = (__bf16)se[(size_t)(kbase + i) * N + n];
    *(v16bf*)(dst + (size_t)tid * 16) = v;
}

// ---------------------------------------------------------------------------
// gating: one wave per token; writes per-expert entry list (entry id = 2t+k)
// and per-entry renormalized weight.
// ---------------------------------------------------------------------------
__global__ __launch_bounds__(TPB)
void moe_gate_kernel(const float* __restrict__ x, const float* __restrict__ gate_w,
                     int* __restrict__ counts, int* __restrict__ ent_list,
                     float* __restrict__ wt_ent) {
    const int lane = threadIdx.x & 31;
    const int wv   = threadIdx.x >> 5;
    const int t    = blockIdx.x * NWAVES + wv;
    if (t >= NTOK) return;

    float acc[NE];
#pragma unroll
    for (int e = 0; e < NE; ++e) acc[e] = 0.f;

    const float* xr = x + (size_t)t * ND;
    for (int k = lane; k < ND; k += 32) {
        const float xv = xr[k];
        const float* g = gate_w + (size_t)k * NE;
#pragma unroll
        for (int e = 0; e < NE; ++e) acc[e] = fmaf(xv, g[e], acc[e]);
    }
#pragma unroll
    for (int e = 0; e < NE; ++e) {
#pragma unroll
        for (int m = 16; m >= 1; m >>= 1) acc[e] += __shfl_xor(acc[e], m, 32);
    }

    if (lane == 0) {
        float mx = acc[0];
#pragma unroll
        for (int e = 1; e < NE; ++e) mx = fmaxf(mx, acc[e]);
        float p[NE]; float sum = 0.f;
#pragma unroll
        for (int e = 0; e < NE; ++e) { p[e] = __expf(acc[e] - mx); sum += p[e]; }
        const float inv = __builtin_amdgcn_rcpf(sum);
#pragma unroll
        for (int e = 0; e < NE; ++e) p[e] *= inv;

        int i0 = 0;
#pragma unroll
        for (int e = 1; e < NE; ++e) if (p[e] > p[i0]) i0 = e;
        int i1 = (i0 == 0) ? 1 : 0;
#pragma unroll
        for (int e = 0; e < NE; ++e) if (e != i0 && p[e] > p[i1]) i1 = e;

        const float denom = p[i0] + p[i1] + 1e-9f;
        wt_ent[2 * t + 0] = p[i0] / denom;
        wt_ent[2 * t + 1] = p[i1] / denom;

        int s0 = atomicAdd(&counts[i0], 1);
        ent_list[i0 * CAP + s0] = 2 * t + 0;
        int s1 = atomicAdd(&counts[i1], 1);
        ent_list[i1 * CAP + s1] = 2 * t + 1;
    }
}

// ---------------------------------------------------------------------------
// PACKED fused expert kernel: 64-token tile, H in two halves, g in LDS,
// contributions stored (not atomics) to dense per-entry buffer.
// ---------------------------------------------------------------------------
__global__ __launch_bounds__(TPB)
void moe_expert_packed(const __bf16* __restrict__ xb,
                       const __bf16* __restrict__ w1p, const __bf16* __restrict__ w2p,
                       const __bf16* __restrict__ w3p,
                       const int* __restrict__ counts, const int* __restrict__ ent_list,
                       float* __restrict__ contrib) {
    const int e   = blockIdx.y;
    const int cnt = counts[e];
    const int t0  = blockIdx.x * MT;
    if (t0 >= cnt) return;

    extern __shared__ char smem[];
    __bf16* sA   = (__bf16*)smem;                          // [MT][LDA2]
    __bf16* sG   = (__bf16*)(smem + SA2_BYTES);            // [MT][LDG2] (half of H)
    int*    entS = (int*)  (smem + SA2_BYTES + SG2_BYTES); // [MT]

    const int lane = threadIdx.x & 31;
    const int wv   = threadIdx.x >> 5;
    const int hi   = lane >> 4;
    const int ln   = lane & 15;

    const __bf16* w1pe = w1p + (size_t)e * ND * NH;
    const __bf16* w2pe = w2p + (size_t)e * ND * NH;
    const __bf16* w3pe = w3p + (size_t)e * NH * ND;

    for (int r = threadIdx.x; r < MT; r += TPB) {
        const int slot = t0 + r;
        entS[r] = (slot < cnt) ? ent_list[e * CAP + slot] : -1;
    }
    __syncthreads();

    // gather x rows (bf16, 16B chunks)
    for (int idx = threadIdx.x; idx < MT * (ND / 8); idx += TPB) {
        const int r = idx >> 7;
        const int c = (idx & 127) * 8;
        const int ent = entS[r];
        const int tok = (ent >= 0) ? (ent >> 1) : 0;
        *(uint4*)&sA[r * LDA2 + c] = *(const uint4*)&xb[(size_t)tok * ND + c];
    }
    __syncthreads();

    for (int hh = 0; hh < 2; ++hh) {
        // ---- GEMM1 half: g[:, hh*1024 .. +1024] -> sG ----
        for (int strip = wv; strip < HHALF / 16; strip += NWAVES) {
            v8f acc1[4] = {}, acc2[4] = {};
            for (int k0 = 0; k0 < ND; k0 += 32) {
                v16bf a[4], b1, b2;
#pragma unroll
                for (int mb = 0; mb < 4; ++mb) {
#pragma unroll
                    for (int i = 0; i < 16; ++i) {
                        const int j  = i >> 1;
                        const int ka = hi * 16 + i;
                        a[mb][i] = sA[(mb * 16 + ln) * LDA2 + k0 + ka];
                    }
                }
                const size_t tb =
                    ((size_t)((hh * (HHALF / 16) + strip) * (ND / 32) + (k0 >> 5)) * 32 + lane) * 16;
                b1 = *(const v16bf*)(w1pe + tb);
                b2 = *(const v16bf*)(w2pe + tb);
#pragma unroll
                for (int mb = 0; mb < 4; ++mb) {
                    acc1[mb] = wmma_bf16(a[mb], b1, acc1[mb]);
                    acc2[mb] = wmma_bf16(a[mb], b2, acc2[mb]);
                }
            }
#pragma unroll
            for (int mb = 0; mb < 4; ++mb) {
#pragma unroll
                for (int r = 0; r < 8; ++r) {
                    const int row = mb * 16 + r + hi * 8;
                    sG[row * LDG2 + strip * 16 + ln] = (__bf16)silu_mul(acc1[mb][r], acc2[mb][r]);
                }
            }
        }
        __syncthreads();

        // ---- GEMM2 half: contrib[entry, :] (+)= g_half @ w3[hh*1024.., :] ----
        for (int strip = wv; strip < ND / 16; strip += NWAVES) {
            const int d0 = strip * 16;
            v8f o[4] = {};
            for (int k0 = 0; k0 < HHALF; k0 += 32) {
                v16bf a[4], b3;
#pragma unroll
                for (int mb = 0; mb < 4; ++mb) {
#pragma unroll
                    for (int i = 0; i < 16; ++i) {
                        const int j  = i >> 1;
                        const int ka = hi * 16 + i;
                        a[mb][i] = sG[(mb * 16 + ln) * LDG2 + k0 + ka];
                    }
                }
                const size_t tb =
                    ((size_t)(strip * (NH / 32) + hh * (HHALF / 32) + (k0 >> 5)) * 32 + lane) * 16;
                b3 = *(const v16bf*)(w3pe + tb);
#pragma unroll
                for (int mb = 0; mb < 4; ++mb) o[mb] = wmma_bf16(a[mb], b3, o[mb]);
            }
#pragma unroll
            for (int mb = 0; mb < 4; ++mb) {
#pragma unroll
                for (int r = 0; r < 8; ++r) {
                    const int row = mb * 16 + r + hi * 8;
                    const int ent = entS[row];
                    if (ent >= 0) {
                        float* p = contrib + (size_t)ent * ND + d0 + ln;
                        if (hh == 0) *p = o[mb][r];
                        else         *p += o[mb][r];   // sole owner: plain RMW is safe
                    }
                }
            }
        }
        __syncthreads();   // before next half overwrites sG
    }
}

// ---------------------------------------------------------------------------
// combine: out[t] = w(2t)*contrib[2t] + w(2t+1)*contrib[2t+1]
// ---------------------------------------------------------------------------
__global__ void moe_combine_kernel(const float4* __restrict__ c4,
                                   const float* __restrict__ wt_ent,
                                   float4* __restrict__ out4) {
    const int i = blockIdx.x * blockDim.x + threadIdx.x;
    if (i >= NTOK * (ND / 4)) return;
    const int t = i >> 8;               // ND/4 = 256 float4 per row
    const int c = i & 255;
    const float w0 = wt_ent[2 * t + 0];
    const float w1 = wt_ent[2 * t + 1];
    const float4 c0 = c4[(size_t)(2 * t + 0) * 256 + c];
    const float4 c1 = c4[(size_t)(2 * t + 1) * 256 + c];
    float4 o;
    o.x = w0 * c0.x + w1 * c1.x;  o.y = w0 * c0.y + w1 * c1.y;
    o.z = w0 * c0.z + w1 * c1.z;  o.w = w0 * c0.w + w1 * c1.w;
    out4[i] = o;
}

// ---------------------------------------------------------------------------
// FALLBACK fused expert kernel (small ws): direct fp32 loads + atomics.
// ---------------------------------------------------------------------------
__global__ __launch_bounds__(TPB)
void moe_expert_fallback(const float* __restrict__ x,
                         const float* __restrict__ w1, const float* __restrict__ w2,
                         const float* __restrict__ w3,
                         const int* __restrict__ counts, const int* __restrict__ ent_list,
                         const float* __restrict__ wt_ent,
                         float* __restrict__ out) {
    const int e   = blockIdx.y;
    const int cnt = counts[e];
    const int t0  = blockIdx.x * MTF;
    if (t0 >= cnt) return;

    extern __shared__ char smem[];
    __bf16* sA   = (__bf16*)smem;
    __bf16* sG   = (__bf16*)(smem + SAF_BYTES);
    int*    tokS = (int*)  (smem + SAF_BYTES + SGF_BYTES);
    float*  wgtS = (float*)(smem + SAF_BYTES + SGF_BYTES + MTF * 4);

    const int lane = threadIdx.x & 31;
    const int wv   = threadIdx.x >> 5;
    const int hi   = lane >> 4;
    const int ln   = lane & 15;

    const float* w1e = w1 + (size_t)e * ND * NH;
    const float* w2e = w2 + (size_t)e * ND * NH;
    const float* w3e = w3 + (size_t)e * NH * ND;

    for (int r = threadIdx.x; r < MTF; r += TPB) {
        const int slot = t0 + r;
        int tk = 0; float w = 0.f;
        if (slot < cnt) { const int ent = ent_list[e * CAP + slot]; tk = ent >> 1; w = wt_ent[ent]; }
        tokS[r] = tk; wgtS[r] = w;
    }
    __syncthreads();
    for (int idx = threadIdx.x; idx < MTF * ND; idx += TPB) {
        const int r = idx >> 10, c = idx & (ND - 1);
        sA[r * LDAF + c] = (__bf16)x[(size_t)tokS[r] * ND + c];
    }
    __syncthreads();

    for (int strip = wv; strip < NH / 16; strip += NWAVES) {
        const int h0 = strip * 16;
        v8f acc1a = {}, acc1b = {}, acc2a = {}, acc2b = {};
        for (int k0 = 0; k0 < ND; k0 += 32) {
            v16bf a0, a1, b1, b2;
#pragma unroll
            for (int i = 0; i < 16; ++i) {
                const int j  = i >> 1;
                const int ka = hi * 16 + i;
                a0[i] = sA[ln * LDAF + k0 + ka];
                a1[i] = sA[(16 + ln) * LDAF + k0 + ka];
                const int kb = hi * 16 + i;
                b1[i] = (__bf16)w1e[(size_t)(k0 + kb) * NH + h0 + ln];
                b2[i] = (__bf16)w2e[(size_t)(k0 + kb) * NH + h0 + ln];
            }
            acc1a = wmma_bf16(a0, b1, acc1a);
            acc1b = wmma_bf16(a1, b1, acc1b);
            acc2a = wmma_bf16(a0, b2, acc2a);
            acc2b = wmma_bf16(a1, b2, acc2b);
        }
#pragma unroll
        for (int r = 0; r < 8; ++r) {
            const int row0 = r + hi * 8;
            sG[row0 * LDGF + h0 + ln] = (__bf16)silu_mul(acc1a[r], acc2a[r]);
            const int row1 = 16 + row0;
            sG[row1 * LDGF + h0 + ln] = (__bf16)silu_mul(acc1b[r], acc2b[r]);
        }
    }
    __syncthreads();

    for (int strip = wv; strip < ND / 16; strip += NWAVES) {
        const int d0 = strip * 16;
        v8f oa = {}, ob = {};
        for (int k0 = 0; k0 < NH; k0 += 32) {
            v16bf a0, a1, b3;
#pragma unroll
            for (int i = 0; i < 16; ++i) {
                const int j  = i >> 1;
                const int ka = hi * 16 + i;
                a0[i] = sG[ln * LDGF + k0 + ka];
                a1[i] = sG[(16 + ln) * LDGF + k0 + ka];
                const int kb = hi * 16 + i;
                b3[i] = (__bf16)w3e[(size_t)(k0 + kb) * ND + d0 + ln];
            }
            oa = wmma_bf16(a0, b3, oa);
            ob = wmma_bf16(a1, b3, ob);
        }
#pragma unroll
        for (int r = 0; r < 8; ++r) {
            const int row0 = r + hi * 8;
            if (wgtS[row0] != 0.f)
                atomicAdd(&out[(size_t)tokS[row0] * ND + d0 + ln], oa[r] * wgtS[row0]);
            const int row1 = 16 + row0;
            if (wgtS[row1] != 0.f)
                atomicAdd(&out[(size_t)tokS[row1] * ND + d0 + ln], ob[r] * wgtS[row1]);
        }
    }
}

// ---------------------------------------------------------------------------
// Host entry
// ---------------------------------------------------------------------------
extern "C" void kernel_launch(void* const* d_in, const int* in_sizes, int n_in,
                              void* d_out, int out_size, void* d_ws, size_t ws_size,
                              hipStream_t stream) {
    const float* x      = (const float*)d_in[0];
    const float* gate_w = (const float*)d_in[1];
    const float* w1     = (const float*)d_in[2];
    const float* w2     = (const float*)d_in[3];
    const float* w3     = (const float*)d_in[4];
    float*       out    = (float*)d_out;

    char* ws = (char*)d_ws;
    int*    counts   = (int*)   (ws + OFF_COUNTS);
    int*    ent_list = (int*)   (ws + OFF_ENT);
    float*  wt_ent   = (float*) (ws + OFF_WTE);
    __bf16* xb       = (__bf16*)(ws + OFF_XB);
    __bf16* w1p      = (__bf16*)(ws + OFF_W1P);
    __bf16* w2p      = (__bf16*)(ws + OFF_W2P);
    __bf16* w3p      = (__bf16*)(ws + OFF_W3P);
    float*  contrib  = (float*) (ws + OFF_CONTRIB);

    const bool packed = (ws_size >= WS_NEEDED);   // fixed per harness -> deterministic

    const int n4 = (NTOK * ND) / 4;
    moe_init_kernel<<<(n4 + TPB - 1) / TPB, TPB, 0, stream>>>((float4*)out, counts, n4);
    moe_gate_kernel<<<NTOK / NWAVES, TPB, 0, stream>>>(x, gate_w, counts, ent_list, wt_ent);

    if (packed) {
        pack_x_kernel<<<(n4 + TPB - 1) / TPB, TPB, 0, stream>>>((const float4*)x, xb);
        const int nthB = NE * (NH / 16) * (ND / 32) * 32;   // == NE*(ND/16)*(NH/32)*32
        pack_b_kernel<<<(nthB + TPB - 1) / TPB, TPB, 0, stream>>>(w1, w1p, ND, NH);
        pack_b_kernel<<<(nthB + TPB - 1) / TPB, TPB, 0, stream>>>(w2, w2p, ND, NH);
        pack_b_kernel<<<(nthB + TPB - 1) / TPB, TPB, 0, stream>>>(w3, w3p, NH, ND);

        dim3 grid(CAP / MT, NE, 1);
        moe_expert_packed<<<grid, TPB, SMEM2, stream>>>(
            xb, w1p, w2p, w3p, counts, ent_list, contrib);

        moe_combine_kernel<<<(n4 + TPB - 1) / TPB, TPB, 0, stream>>>(
            (const float4*)contrib, wt_ent, (float4*)out);
    } else {
        dim3 grid(CAP / MTF, NE, 1);
        moe_expert_fallback<<<grid, TPB, SMEMF, stream>>>(
            x, w1, w2, w3, counts, ent_list, wt_ent, out);
    }
}
